// LayerGIN_1151051235411
// MI455X (gfx1250) — matrix operand order repaired and checked by reading, not compiled, not run on hardware
//
#include <hip/hip_runtime.h>

#define DIN 128
#define DHID 512
#define DOUT 128

typedef __attribute__((ext_vector_type(16))) __bf16 v16bf;
typedef __attribute__((ext_vector_type(8)))  float  v8f;

// ---------------------------------------------------------------- utilities

__global__ void zero_kernel(float* p, long long n) {
  long long i = (long long)blockIdx.x * blockDim.x + threadIdx.x;
  long long stride = (long long)gridDim.x * blockDim.x;
  for (; i < n; i += stride) p[i] = 0.0f;
}

// one wave (32 lanes) per edge: float4 gather of v[src], 4 f32 atomics into v_agg[dst]
__global__ __launch_bounds__(256) void scatter_kernel(
    const float* __restrict__ v, const int* __restrict__ src,
    const int* __restrict__ dst, const float* __restrict__ w,
    float* __restrict__ vagg, int E) {
  int e = blockIdx.x * 8 + (threadIdx.x >> 5);
  if (e >= E) return;
  int lane = threadIdx.x & 31;
  int s = src[e];
  int d = dst[e];
  float ww = w[e];
  const float4 x = *(const float4*)(v + (size_t)s * DIN + lane * 4);
  float* o = vagg + (size_t)d * DIN + lane * 4;
  atomicAdd(o + 0, ww * x.x);
  atomicAdd(o + 1, ww * x.y);
  atomicAdd(o + 2, ww * x.z);
  atomicAdd(o + 3, ww * x.w);
}

__global__ void epsadd_kernel(float* __restrict__ vagg, const float* __restrict__ v,
                              const float* __restrict__ eps, long long n) {
  long long i = (long long)blockIdx.x * blockDim.x + threadIdx.x;
  if (i < n) vagg[i] = fmaf(eps[0], v[i], vagg[i]);
}

// ------------------------------------------------- weight packing (B-fragments)
// B fragment for V_WMMA_F32_16X16X32_BF16 (ISA 7.12.2): 32x16 tile, lane = column
// (lane&15) of tile, khalf = lane>>4: elems [0..7] = K 8*khalf..+7,
// [8..15] = K 16+8*khalf..+7. One wave packs one fragment; each lane
// writes its 16 bf16 (32B) contiguously -> GEMM loads are 2x b128 per lane.
__global__ __launch_bounds__(256) void packB_kernel(const float* __restrict__ W,
                                                    __bf16* __restrict__ Bp,
                                                    int K, int C) {
  int f = blockIdx.x * 8 + (threadIdx.x >> 5);
  int NT = C >> 4;
  int nfrag = (K >> 5) * NT;
  if (f >= nfrag) return;
  int lane = threadIdx.x & 31;
  int kt = f / NT, nt = f % NT;
  int n = nt * 16 + (lane & 15);
  int kb = kt * 32 + (lane >> 4) * 8;
  v16bf frag;
#pragma unroll
  for (int j = 0; j < 16; ++j) frag[j] = (__bf16)W[(size_t)(kb + (j < 8 ? j : j + 8)) * C + n];
  *(v16bf*)(Bp + (size_t)f * 512 + (size_t)lane * 16) = frag;
}

// ------------------------------------------------------------------ WMMA GEMM
// A fragment (16x32 bf16, ISA 7.12.2): lane holds row (lane&15); with
// khalf = lane>>4, vector elems [0..7] = A[row][kb..kb+7], [8..15] =
// A[row][kb+16..kb+23], kb = kt*32 + 8*khalf  -> two contiguous 32B f32 spans.
template <bool BN>
__device__ inline v16bf load_a_frag(const float* __restrict__ A, long long rowoff, int kb,
                                    const float* __restrict__ scale,
                                    const float* __restrict__ shift) {
  const float* p = A + rowoff + kb;
  float4 a0 = *(const float4*)(p);
  float4 a1 = *(const float4*)(p + 4);
  float4 a2 = *(const float4*)(p + 16);
  float4 a3 = *(const float4*)(p + 20);
  float x[16] = {a0.x, a0.y, a0.z, a0.w, a1.x, a1.y, a1.z, a1.w,
                 a2.x, a2.y, a2.z, a2.w, a3.x, a3.y, a3.z, a3.w};
  if (BN) {
    float4 s0 = *(const float4*)(scale + kb);
    float4 s1 = *(const float4*)(scale + kb + 4);
    float4 s2 = *(const float4*)(scale + kb + 16);
    float4 s3 = *(const float4*)(scale + kb + 20);
    float4 t0 = *(const float4*)(shift + kb);
    float4 t1 = *(const float4*)(shift + kb + 4);
    float4 t2 = *(const float4*)(shift + kb + 16);
    float4 t3 = *(const float4*)(shift + kb + 20);
    float sc[16] = {s0.x, s0.y, s0.z, s0.w, s1.x, s1.y, s1.z, s1.w,
                    s2.x, s2.y, s2.z, s2.w, s3.x, s3.y, s3.z, s3.w};
    float sh[16] = {t0.x, t0.y, t0.z, t0.w, t1.x, t1.y, t1.z, t1.w,
                    t2.x, t2.y, t2.z, t2.w, t3.x, t3.y, t3.z, t3.w};
#pragma unroll
    for (int i = 0; i < 16; ++i) x[i] = fmaxf(fmaf(x[i], sc[i], sh[i]), 0.0f);
  }
  v16bf r;
#pragma unroll
  for (int i = 0; i < 16; ++i) r[i] = (__bf16)x[i];
  return r;
}

template <int NDIM>
__device__ inline void store_tile(float* __restrict__ Out, v8f acc, int mbase, int n, float bj) {
#pragma unroll
  for (int r = 0; r < 8; ++r)
    Out[(size_t)(mbase + r) * NDIM + n] = acc[r] + bj;
}

// Block = 256 threads = 8 waves. Wave w computes rows [16*(8*bx+w)) x 64 cols
// (4 n-tiles, A fragment reused 4x in registers). M must be padded to 128.
template <int KDIM, int NDIM, bool BN>
__global__ __launch_bounds__(256) void gemm_kernel(
    const float* __restrict__ A, const __bf16* __restrict__ Bp,
    const float* __restrict__ bias, const float* __restrict__ scale,
    const float* __restrict__ shift, float* __restrict__ Out) {
  const int lane = threadIdx.x & 31;
  const int wave = threadIdx.x >> 5;
  const int mtile = blockIdx.x * 8 + wave;
  const int khalf = lane >> 4;
  const long long rowoff = (long long)(mtile * 16 + (lane & 15)) * KDIM;
  constexpr int NT = NDIM / 16;
  constexpr int KT = KDIM / 32;
  const int nt0 = blockIdx.y * 4;

  v8f acc0 = {}, acc1 = {}, acc2 = {}, acc3 = {};
  for (int kt = 0; kt < KT; ++kt) {
    const int kb = kt * 32 + khalf * 8;
    v16bf a = load_a_frag<BN>(A, rowoff, kb, scale, shift);
    const __bf16* bb = Bp + ((size_t)(kt * NT + nt0) * 32 + (size_t)lane) * 16;
    v16bf b0 = *(const v16bf*)(bb);
    v16bf b1 = *(const v16bf*)(bb + 512);
    v16bf b2 = *(const v16bf*)(bb + 1024);
    v16bf b3 = *(const v16bf*)(bb + 1536);
    acc0 = __builtin_amdgcn_wmma_f32_16x16x32_bf16(false, a, false, b0, (short)0, acc0, false, false);
    acc1 = __builtin_amdgcn_wmma_f32_16x16x32_bf16(false, a, false, b1, (short)0, acc1, false, false);
    acc2 = __builtin_amdgcn_wmma_f32_16x16x32_bf16(false, a, false, b2, (short)0, acc2, false, false);
    acc3 = __builtin_amdgcn_wmma_f32_16x16x32_bf16(false, a, false, b3, (short)0, acc3, false, false);
  }
  // C/D layout: lane holds column (lane&15), rows khalf*8 + r (ISA 7.12.2)
  const int mbase = mtile * 16 + khalf * 8;
  const int ncol = lane & 15;
  {
    int n = (nt0 + 0) * 16 + ncol;
    store_tile<NDIM>(Out, acc0, mbase, n, bias[n]);
  }
  {
    int n = (nt0 + 1) * 16 + ncol;
    store_tile<NDIM>(Out, acc1, mbase, n, bias[n]);
  }
  {
    int n = (nt0 + 2) * 16 + ncol;
    store_tile<NDIM>(Out, acc2, mbase, n, bias[n]);
  }
  {
    int n = (nt0 + 3) * 16 + ncol;
    store_tile<NDIM>(Out, acc3, mbase, n, bias[n]);
  }
}

// ---------------------------------------------------------------- BN helpers

// column sum / sum-of-squares. Block = 256 threads = 64 columns x 4 row-groups,
// each block covers 1024 rows; LDS reduce, then 2 atomics per column per block.
__global__ __launch_bounds__(256) void colstats_kernel(
    const float* __restrict__ X, int nrows, int C,
    float* __restrict__ sum, float* __restrict__ sumsq) {
  const int c_local = threadIdx.x & 63;
  const int rg = threadIdx.x >> 6;
  const int col = blockIdx.x * 64 + c_local;
  int r = blockIdx.y * 1024 + rg;
  float s = 0.0f, ss = 0.0f;
  for (int i = 0; i < 256; ++i, r += 4) {
    if (r < nrows) {
      float x = X[(size_t)r * C + col];
      s += x;
      ss += x * x;
    }
  }
  __shared__ float sh_s[256];
  __shared__ float sh_q[256];
  sh_s[threadIdx.x] = s;
  sh_q[threadIdx.x] = ss;
  __syncthreads();
  if (rg == 0) {
    s = sh_s[c_local] + sh_s[c_local + 64] + sh_s[c_local + 128] + sh_s[c_local + 192];
    ss = sh_q[c_local] + sh_q[c_local + 64] + sh_q[c_local + 128] + sh_q[c_local + 192];
    atomicAdd(sum + col, s);
    atomicAdd(sumsq + col, ss);
  }
}

// fold mean/var/gamma/beta into per-channel scale & shift
__global__ void bnparams_kernel(const float* __restrict__ sum, const float* __restrict__ sumsq,
                                const float* __restrict__ gamma, const float* __restrict__ beta,
                                float* __restrict__ scale, float* __restrict__ shift,
                                int C, float inv_n) {
  int c = blockIdx.x * blockDim.x + threadIdx.x;
  if (c >= C) return;
  float mean = sum[c] * inv_n;
  float var = sumsq[c] * inv_n - mean * mean;
  float is = rsqrtf(var + 1e-5f);
  float sc = gamma[c] * is;
  scale[c] = sc;
  shift[c] = beta[c] - mean * sc;
}

__global__ void bnrelu_kernel(const float* __restrict__ X, const float* __restrict__ scale,
                              const float* __restrict__ shift, float* __restrict__ out,
                              long long n, int cmask) {
  long long i = (long long)blockIdx.x * blockDim.x + threadIdx.x;
  if (i >= n) return;
  int c = (int)(i & cmask);
  out[i] = fmaxf(fmaf(X[i], scale[c], shift[c]), 0.0f);
}

// ------------------------------------------------------------------- launcher

extern "C" void kernel_launch(void* const* d_in, const int* in_sizes, int n_in,
                              void* d_out, int out_size, void* d_ws, size_t ws_size,
                              hipStream_t stream) {
  const float* v   = (const float*)d_in[0];
  const int*   src = (const int*)d_in[1];
  const int*   dst = (const int*)d_in[2];
  const float* ew  = (const float*)d_in[3];
  const float* eps = (const float*)d_in[4];
  const float* W1  = (const float*)d_in[5];
  const float* b1  = (const float*)d_in[6];
  const float* g1  = (const float*)d_in[7];
  const float* be1 = (const float*)d_in[8];
  const float* W2  = (const float*)d_in[9];
  const float* b2  = (const float*)d_in[10];
  const float* g2  = (const float*)d_in[11];
  const float* be2 = (const float*)d_in[12];

  const int N = in_sizes[0] / DIN;        // 50000 nodes
  const int E = in_sizes[1];              // 800000 edges
  const int M_PAD = ((N + 127) / 128) * 128;  // 50048 (N itself is 16-aligned)

  // workspace layout (floats)
  float* ws   = (float*)d_ws;
  float* vagg = ws;                                   // M_PAD*128 (reused as OUT_pre)
  float* Hpre = vagg + (size_t)M_PAD * DIN;           // M_PAD*512
  float* bp1f = Hpre + (size_t)M_PAD * DHID;
  __bf16* Bp1 = (__bf16*)bp1f;                        // 128*512 bf16
  float* bp2f = bp1f + (DIN * DHID) / 2;
  __bf16* Bp2 = (__bf16*)bp2f;                        // 512*128 bf16
  float* stats = bp2f + (DHID * DOUT) / 2;
  float* sum1 = stats;
  float* sq1  = sum1 + DHID;
  float* sc1  = sq1 + DHID;
  float* sh1  = sc1 + DHID;
  float* sum2 = sh1 + DHID;
  float* sq2  = sum2 + DOUT;
  float* sc2  = sq2 + DOUT;
  float* sh2  = sc2 + DOUT;

  // zero accumulator (incl. padded rows) and stats sums
  zero_kernel<<<2048, 256, 0, stream>>>(vagg, (long long)M_PAD * DIN);
  zero_kernel<<<8, 256, 0, stream>>>(stats, (long long)(4 * DHID + 4 * DOUT));

  // 1) edge scatter: v_agg[dst] += w * v[src]   (L2-resident atomics)
  scatter_kernel<<<(E + 7) / 8, 256, 0, stream>>>(v, src, dst, ew, vagg, E);

  // 2) epsilon residual (valid rows only; padded rows stay zero)
  {
    long long n = (long long)N * DIN;
    epsadd_kernel<<<(unsigned)((n + 255) / 256), 256, 0, stream>>>(vagg, v, eps, n);
  }

  // pack W1, W2 into WMMA B-fragment layout (bf16)
  packB_kernel<<<16, 256, 0, stream>>>(W1, Bp1, DIN, DHID);   // 4*32 = 128 frags
  packB_kernel<<<16, 256, 0, stream>>>(W2, Bp2, DHID, DOUT);  // 16*8 = 128 frags

  // 3) GEMM1: Hpre = vagg @ W1 + b1
  {
    dim3 g(M_PAD / 128, DHID / 64);
    gemm_kernel<DIN, DHID, false><<<g, 256, 0, stream>>>(vagg, Bp1, b1, nullptr, nullptr, Hpre);
  }
  // BN1 stats over valid rows only
  {
    dim3 g(DHID / 64, (N + 1023) / 1024);
    colstats_kernel<<<g, 256, 0, stream>>>(Hpre, N, DHID, sum1, sq1);
    bnparams_kernel<<<1, DHID, 0, stream>>>(sum1, sq1, g1, be1, sc1, sh1, DHID, 1.0f / (float)N);
  }

  // 4) GEMM2: OUTpre = relu(BN1(Hpre)) @ W2 + b2   (BN+ReLU fused into A load)
  float* OUTpre = vagg;  // vagg no longer needed
  {
    dim3 g(M_PAD / 128, DOUT / 64);
    gemm_kernel<DHID, DOUT, true><<<g, 256, 0, stream>>>(Hpre, Bp2, b2, sc1, sh1, OUTpre);
  }
  // BN2 stats
  {
    dim3 g(DOUT / 64, (N + 1023) / 1024);
    colstats_kernel<<<g, 256, 0, stream>>>(OUTpre, N, DOUT, sum2, sq2);
    bnparams_kernel<<<1, DOUT, 0, stream>>>(sum2, sq2, g2, be2, sc2, sh2, DOUT, 1.0f / (float)N);
  }

  // 5) final BN2 + ReLU into d_out (valid rows only)
  {
    long long n = (long long)N * DOUT;
    bnrelu_kernel<<<(unsigned)((n + 255) / 256), 256, 0, stream>>>(
        OUTpre, sc2, sh2, (float*)d_out, n, DOUT - 1);
  }
}
